// HAN_39238821216884
// MI455X (gfx1250) — compile-verified, matrix operand order checked by reading, not run on hardware
//
#include <hip/hip_runtime.h>
#include <cstdint>
#include <math.h>

// ---------------------------------------------------------------------------
// Problem constants (match reference)
// ---------------------------------------------------------------------------
#define HN   8
#define DHN  32
#define FD   256
#define NPn  100000
#define NAn  50000
#define EAPn 300000
#define EPAn 300000
#define EPPn 500000
#define NCLS 32

typedef _Float16 v8h  __attribute__((ext_vector_type(8)));
typedef _Float16 v16h __attribute__((ext_vector_type(16)));
typedef float    v8f  __attribute__((ext_vector_type(8)));

// ---------------------------------------------------------------------------
// WMMA GEMM:  out[M,N] = A[M,K] @ W[K,N] + bias   (f32 in, f16 WMMA, f32 acc)
//  mode 0: store out (+bias), guarded by M/N
//  mode 1: colsum[col] += sum_over_valid_rows tanh(acc + bias[col])
//  reluA : apply relu to A elements on load (fused relu of previous stage)
// Block: 256 threads (8 waves). Block tile 128x128, K-step 32.
// Wave w: (w>>1) selects 32-row strip, (w&1) selects 64-col strip; 2x4 WMMA.
// ---------------------------------------------------------------------------
#define BM 128
#define BN 128
#define BK 32
#define LDT 40   // padded LDS row length in halves (80B -> conflict-free-ish)

__global__ void __launch_bounds__(256)
gemm_wmma_f16(const float* __restrict__ A, const float* __restrict__ W,
              const float* __restrict__ bias, float* __restrict__ out,
              float* __restrict__ colsum,
              int M, int N, int K, int reluA, int mode)
{
    __shared__ __align__(16) _Float16 Ah[BM * LDT];
    __shared__ __align__(16) _Float16 Bt[BN * LDT];   // transposed: Bt[n][k]

    const int tid  = threadIdx.x;
    const int lane = tid & 31;
    const int wid  = tid >> 5;
    const int wm   = wid >> 1;          // 0..3  -> 32-row strip
    const int wn   = wid & 1;           // 0..1  -> 64-col strip
    const long tileM = (long)blockIdx.x * BM;
    const long tileN = (long)blockIdx.y * BN;

    v8f acc[2][4] = {};

    for (int k0 = 0; k0 < K; k0 += BK) {
        // ---- stage A tile (128x32 f32 -> f16), 4 float4 per thread ----
        {
            int r0 = tid >> 3;
            int c  = (tid & 7) << 2;
            #pragma unroll
            for (int rr = 0; rr < 4; ++rr) {
                int  r    = r0 + rr * 32;
                long grow = tileM + r;
                float4 v = make_float4(0.f, 0.f, 0.f, 0.f);
                if (grow < M) v = *(const float4*)(A + grow * (long)K + k0 + c);
                if (reluA) {
                    v.x = fmaxf(v.x, 0.f); v.y = fmaxf(v.y, 0.f);
                    v.z = fmaxf(v.z, 0.f); v.w = fmaxf(v.w, 0.f);
                }
                _Float16* d = &Ah[r * LDT + c];
                d[0] = (_Float16)v.x; d[1] = (_Float16)v.y;
                d[2] = (_Float16)v.z; d[3] = (_Float16)v.w;
            }
        }
        // ---- stage B tile transposed (32x128 -> Bt[128][32]), 4 float4/thr ----
        {
            #pragma unroll
            for (int qq = 0; qq < 4; ++qq) {
                int q   = tid + qq * 256;
                int k   = q >> 5;
                int col = (q & 31) << 2;
                float4 v = make_float4(0.f, 0.f, 0.f, 0.f);
                if (tileN + col + 3 < N)
                    v = *(const float4*)(W + (long)(k0 + k) * N + tileN + col);
                Bt[(col + 0) * LDT + k] = (_Float16)v.x;
                Bt[(col + 1) * LDT + k] = (_Float16)v.y;
                Bt[(col + 2) * LDT + k] = (_Float16)v.z;
                Bt[(col + 3) * LDT + k] = (_Float16)v.w;
            }
        }
        __syncthreads();

        // ---- fragments + WMMA ----
        const int khalf = lane >> 4;
        const int mrow  = lane & 15;
        v16h af[2], bf[4];
        #pragma unroll
        for (int tm = 0; tm < 2; ++tm) {
            const _Float16* row = &Ah[(wm * 32 + tm * 16 + mrow) * LDT];
            v8h lo = *(const v8h*)(row + khalf * 8);
            v8h hi = *(const v8h*)(row + 16 + khalf * 8);
            #pragma unroll
            for (int i = 0; i < 8; ++i) { af[tm][i] = lo[i]; af[tm][8 + i] = hi[i]; }
        }
        #pragma unroll
        for (int tn = 0; tn < 4; ++tn) {
            const _Float16* row = &Bt[(wn * 64 + tn * 16 + mrow) * LDT];
            v8h lo = *(const v8h*)(row + khalf * 8);
            v8h hi = *(const v8h*)(row + 16 + khalf * 8);
            #pragma unroll
            for (int i = 0; i < 8; ++i) { bf[tn][i] = lo[i]; bf[tn][8 + i] = hi[i]; }
        }
        #pragma unroll
        for (int tm = 0; tm < 2; ++tm)
            #pragma unroll
            for (int tn = 0; tn < 4; ++tn)
                acc[tm][tn] = __builtin_amdgcn_wmma_f32_16x16x32_f16(
                    false, af[tm], false, bf[tn], (short)0, acc[tm][tn],
                    false, false);
        __syncthreads();
    }

    // ---- epilogue ----
    const int mrow = lane & 15;
    const int hi16 = lane >> 4;     // 0 or 1
    if (mode == 0) {
        #pragma unroll
        for (int tm = 0; tm < 2; ++tm)
            #pragma unroll
            for (int tn = 0; tn < 4; ++tn) {
                long col = tileN + wn * 64 + tn * 16 + mrow;
                if (col >= N) continue;
                float b = bias ? bias[col] : 0.f;
                #pragma unroll
                for (int v = 0; v < 8; ++v) {
                    long row = tileM + wm * 32 + tm * 16 + hi16 * 8 + v;
                    if (row < M) out[row * (long)N + col] = acc[tm][tn][v] + b;
                }
            }
    } else {
        #pragma unroll
        for (int tn = 0; tn < 4; ++tn) {
            long col = tileN + wn * 64 + tn * 16 + mrow;
            float b  = bias ? bias[col] : 0.f;
            float cs = 0.f;
            #pragma unroll
            for (int tm = 0; tm < 2; ++tm)
                #pragma unroll
                for (int v = 0; v < 8; ++v) {
                    long row = tileM + wm * 32 + tm * 16 + hi16 * 8 + v;
                    if (row < M) cs += tanhf(acc[tm][tn][v] + b);
                }
            cs += __shfl_xor(cs, 16, 32);
            if (hi16 == 0 && col < N) atomicAdd(&colsum[col], cs);
        }
    }
}

// ---------------------------------------------------------------------------
// Utility / graph kernels
// ---------------------------------------------------------------------------
__global__ void fill_u32(unsigned* __restrict__ p, unsigned v, long n) {
    long i = (long)blockIdx.x * 256 + threadIdx.x;
    if (i < n) p[i] = v;
}

// per-node, per-head dot:  logit[n,h] = sum_d h[n,h,d]*a[h,d]
__global__ void __launch_bounds__(256)
node_logits(const float* __restrict__ h, const float* __restrict__ avec,
            float* __restrict__ out, int Nn)
{
    int n = blockIdx.x;
    if (n >= Nn) return;
    int t = threadIdx.x;                    // 256 = 8 heads * 32 dims
    float v = h[(long)n * FD + t] * avec[t];
    #pragma unroll
    for (int o = 16; o > 0; o >>= 1) v += __shfl_down(v, o, 32);
    if ((t & 31) == 0) out[(long)n * HN + (t >> 5)] = v;
}

__device__ inline unsigned fkey(float f) {
    unsigned u = __float_as_uint(f);
    return (u & 0x80000000u) ? ~u : (u | 0x80000000u);
}
__device__ inline float funkey(unsigned k) {
    unsigned u = (k & 0x80000000u) ? (k & 0x7FFFFFFFu) : ~k;
    return __uint_as_float(u);
}

__global__ void edge_alpha_max(const float* __restrict__ als, const float* __restrict__ ald,
                               const int* __restrict__ src, const int* __restrict__ dst,
                               float* __restrict__ alpha, unsigned* __restrict__ mkey, int E)
{
    long t = (long)blockIdx.x * 256 + threadIdx.x;
    if (t >= (long)E * HN) return;
    int e = (int)(t >> 3), hh = (int)(t & 7);
    float a = als[(long)src[e] * HN + hh] + ald[(long)dst[e] * HN + hh];
    a = (a > 0.f) ? a : 0.2f * a;                  // leaky relu 0.2
    alpha[t] = a;
    atomicMax(&mkey[(long)dst[e] * HN + hh], fkey(a));
}

__global__ void edge_exp_sum(float* __restrict__ alpha, const int* __restrict__ dst,
                             const unsigned* __restrict__ mkey, float* __restrict__ ssum, int E)
{
    long t = (long)blockIdx.x * 256 + threadIdx.x;
    if (t >= (long)E * HN) return;
    int e = (int)(t >> 3), hh = (int)(t & 7);
    float m  = funkey(mkey[(long)dst[e] * HN + hh]);
    float ex = __expf(alpha[t] - m);
    alpha[t] = ex;
    atomicAdd(&ssum[(long)dst[e] * HN + hh], ex);
}

// out[dst,f] += h_src[src,f] * alphaExp[e,h]/(s[dst,h]+eps); one block per edge
__global__ void __launch_bounds__(256)
edge_aggregate(const float* __restrict__ hsrc, const float* __restrict__ alphaExp,
               const float* __restrict__ ssum, const int* __restrict__ src,
               const int* __restrict__ dst, float* __restrict__ out, int E)
{
    int e = blockIdx.x;
    if (e >= E) return;
    int f = threadIdx.x, hh = f >> 5;
    int d = dst[e], s = src[e];
    float w = alphaExp[(long)e * HN + hh] / (ssum[(long)d * HN + hh] + 1e-16f);
    atomicAdd(&out[(long)d * FD + f], hsrc[(long)s * FD + f] * w);
}

// K=2 semantic softmax: score_k = q . (colsum_k * invN); attn = softmax(score)
__global__ void __launch_bounds__(256)
semantic_score(const float* __restrict__ colsum, const float* __restrict__ q,
               float* __restrict__ attn, float invN)
{
    __shared__ float red[256];
    __shared__ float sc[2];
    int t = threadIdx.x;
    for (int k = 0; k < 2; ++k) {
        red[t] = colsum[k * FD + t] * invN * q[t];
        __syncthreads();
        for (int s = 128; s > 0; s >>= 1) {
            if (t < s) red[t] += red[t + s];
            __syncthreads();
        }
        if (t == 0) sc[k] = red[0];
        __syncthreads();
    }
    if (t == 0) {
        float m  = fmaxf(sc[0], sc[1]);
        float e0 = __expf(sc[0] - m), e1 = __expf(sc[1] - m);
        float d  = e0 + e1;
        attn[0] = e0 / d;
        attn[1] = e1 / d;
    }
}

__global__ void combine2(const float* __restrict__ x0, const float* __restrict__ x1,
                         const float* __restrict__ attn, float* __restrict__ out, long n)
{
    long i = (long)blockIdx.x * 256 + threadIdx.x;
    if (i >= n) return;
    out[i] = attn[0] * fmaxf(x0[i], 0.f) + attn[1] * fmaxf(x1[i], 0.f);
}

__global__ void relu_copy(const float* __restrict__ x, float* __restrict__ out, long n) {
    long i = (long)blockIdx.x * 256 + threadIdx.x;
    if (i < n) out[i] = fmaxf(x[i], 0.f);
}

// ---------------------------------------------------------------------------
// Host-side orchestration
// ---------------------------------------------------------------------------
static inline long cdivl(long a, long b) { return (a + b - 1) / b; }

static void launch_gemm(hipStream_t s, const float* A, const float* W, const float* bias,
                        float* out, float* colsum, int M, int N, int K, int reluA, int mode)
{
    dim3 g((unsigned)cdivl(M, BM), (unsigned)cdivl(N, BN));
    gemm_wmma_f16<<<g, 256, 0, s>>>(A, W, bias, out, colsum, M, N, K, reluA, mode);
}

struct Scratch {
    float*    als;   float* ald;   float* alpha;
    unsigned* mkey;  float* ssum;
    float*    colsum; float* attn;
};

static void run_edge_type(hipStream_t s, const float* hs, int Ns, const float* hd, int Nd,
                          const float* aS, const float* aD,
                          const int* src, const int* dst, int E,
                          float* outb, const Scratch& sc)
{
    node_logits<<<Ns, 256, 0, s>>>(hs, aS, sc.als, Ns);
    node_logits<<<Nd, 256, 0, s>>>(hd, aD, sc.ald, Nd);
    long nm = (long)Nd * HN;
    fill_u32<<<(unsigned)cdivl(nm, 256), 256, 0, s>>>(sc.mkey, 0u, nm);
    fill_u32<<<(unsigned)cdivl(nm, 256), 256, 0, s>>>((unsigned*)sc.ssum, 0u, nm);
    long no = (long)Nd * FD;
    fill_u32<<<(unsigned)cdivl(no, 256), 256, 0, s>>>((unsigned*)outb, 0u, no);
    long ne = (long)E * HN;
    edge_alpha_max<<<(unsigned)cdivl(ne, 256), 256, 0, s>>>(sc.als, sc.ald, src, dst, sc.alpha, sc.mkey, E);
    edge_exp_sum  <<<(unsigned)cdivl(ne, 256), 256, 0, s>>>(sc.alpha, dst, sc.mkey, sc.ssum, E);
    edge_aggregate<<<E, 256, 0, s>>>(hs, sc.alpha, sc.ssum, src, dst, outb, E);
}

static void run_layer(hipStream_t s,
                      const float* xP, const float* xA, int Kp, int Ka,
                      const float* WP, const float* bP, const float* WA, const float* bA,
                      const float* asAP, const float* adAP,
                      const float* asPA, const float* adPA,
                      const float* asPP, const float* adPP,
                      const float* kW, const float* kb, const float* q,
                      const int* sAP, const int* dAP, const int* sPA, const int* dPA,
                      const int* sPP, const int* dPP,
                      float* hP, float* hA, float* oAP, float* oPP, float* oPA,
                      float* newP, float* newA, const Scratch& sc)
{
    // projections
    launch_gemm(s, xP, WP, bP, hP, nullptr, NPn, FD, Kp, 0, 0);
    launch_gemm(s, xA, WA, bA, hA, nullptr, NAn, FD, Ka, 0, 0);
    // edge attentions
    run_edge_type(s, hA, NAn, hP, NPn, asAP, adAP, sAP, dAP, EAPn, oAP, sc);
    run_edge_type(s, hP, NPn, hP, NPn, asPP, adPP, sPP, dPP, EPPn, oPP, sc);
    if (oPA) run_edge_type(s, hP, NPn, hA, NAn, asPA, adPA, sPA, dPA, EPAn, oPA, sc);
    // semantic attention for papers (K=2); tanh+colsum fused in GEMM epilogue
    fill_u32<<<2, 256, 0, s>>>((unsigned*)sc.colsum, 0u, 2 * FD);
    launch_gemm(s, oAP, kW, kb, nullptr, sc.colsum,      NPn, FD, FD, 1, 1);
    launch_gemm(s, oPP, kW, kb, nullptr, sc.colsum + FD, NPn, FD, FD, 1, 1);
    semantic_score<<<1, 256, 0, s>>>(sc.colsum, q, sc.attn, 1.f / (float)NPn);
    long np = (long)NPn * FD;
    combine2<<<(unsigned)cdivl(np, 256), 256, 0, s>>>(oAP, oPP, sc.attn, newP, np);
    // authors: K=1 -> semantic attention is identity; newA = relu(oPA)
    if (newA && oPA) {
        long na = (long)NAn * FD;
        relu_copy<<<(unsigned)cdivl(na, 256), 256, 0, s>>>(oPA, newA, na);
    }
}

extern "C" void kernel_launch(void* const* d_in, const int* in_sizes, int n_in,
                              void* d_out, int out_size, void* d_ws, size_t ws_size,
                              hipStream_t stream)
{
    const float* x_P = (const float*)d_in[0];
    const float* x_A = (const float*)d_in[1];
    const int* sAP = (const int*)d_in[2];
    const int* dAP = (const int*)d_in[3];
    const int* sPA = (const int*)d_in[4];
    const int* dPA = (const int*)d_in[5];
    const int* sPP = (const int*)d_in[6];
    const int* dPP = (const int*)d_in[7];
    const float* P0[13], *P1[13];
    for (int i = 0; i < 13; ++i) P0[i] = (const float*)d_in[8 + i];
    for (int i = 0; i < 13; ++i) P1[i] = (const float*)d_in[21 + i];
    const float* linW = (const float*)d_in[34];
    const float* linb = (const float*)d_in[35];
    float* out = (float*)d_out;

    // ---- carve workspace ----
    size_t off = 0;
    auto carve = [&](size_t nbytes) -> void* {
        void* r = (char*)d_ws + off;
        off += (nbytes + 255) & ~(size_t)255;
        return r;
    };
    float* bA_ = (float*)carve((size_t)NPn * FD * 4);  // hP(L0) / newP(L0)=xP(L1) / oPP(L1)
    float* bB_ = (float*)carve((size_t)NAn * FD * 4);  // hA(L0) / newA(L0)=xA(L1)
    float* bC_ = (float*)carve((size_t)NPn * FD * 4);  // oAP(L0) / hP(L1) / newP(L1)
    float* bD_ = (float*)carve((size_t)NPn * FD * 4);  // oPP(L0) / oAP(L1)
    float* bE_ = (float*)carve((size_t)NAn * FD * 4);  // oPA(L0) / hA(L1)
    Scratch sc;
    sc.als    = (float*)carve((size_t)NPn * HN * 4);
    sc.ald    = (float*)carve((size_t)NPn * HN * 4);
    sc.alpha  = (float*)carve((size_t)EPPn * HN * 4);
    sc.mkey   = (unsigned*)carve((size_t)NPn * HN * 4);
    sc.ssum   = (float*)carve((size_t)NPn * HN * 4);
    sc.colsum = (float*)carve(2 * FD * 4);
    sc.attn   = (float*)carve(2 * 4);
    (void)ws_size; (void)in_sizes; (void)n_in; (void)out_size;

    // ---- layer 0: xP[512] xA[256] -> newP=bA_, newA=bB_ ----
    run_layer(stream, x_P, x_A, 512, 256,
              P0[0], P0[1], P0[2], P0[3],         // W_P b_P W_A b_A
              P0[4], P0[5], P0[6], P0[7], P0[8], P0[9],  // as/ad AP, PA, PP
              P0[10], P0[11], P0[12],             // kW kb q
              sAP, dAP, sPA, dPA, sPP, dPP,
              /*hP*/ bA_, /*hA*/ bB_, /*oAP*/ bC_, /*oPP*/ bD_, /*oPA*/ bE_,
              /*newP*/ bA_, /*newA*/ bB_, sc);
    // note: newP written into bA_ only after hP(bA_) is dead (combine reads bC_,bD_)

    // ---- layer 1: xP=bA_, xA=bB_; authors' output unused -> skip PA ----
    run_layer(stream, bA_, bB_, 256, 256,
              P1[0], P1[1], P1[2], P1[3],
              P1[4], P1[5], P1[6], P1[7], P1[8], P1[9],
              P1[10], P1[11], P1[12],
              sAP, dAP, sPA, dPA, sPP, dPP,
              /*hP*/ bC_, /*hA*/ bE_, /*oAP*/ bD_, /*oPP*/ bA_, /*oPA*/ nullptr,
              /*newP*/ bC_, /*newA*/ nullptr, sc);

    // ---- final classifier: out[NP,32] = newP @ lin_W + lin_b ----
    launch_gemm(stream, bC_, linW, linb, out, nullptr, NPn, NCLS, FD, 0, 0);
}
